// EfficientAdditiveAttnetion_75453985457223
// MI455X (gfx1250) — matrix operand order repaired and checked by reading, not compiled, not run on hardware
//
#include <hip/hip_runtime.h>
#include <hip/hip_bf16.h>

typedef __attribute__((ext_vector_type(16))) _Float16 v16h;
typedef __attribute__((ext_vector_type(8)))  _Float16 v8h;
typedef __attribute__((ext_vector_type(8)))  float    v8f;

#define BATCH 32
#define NTOK  1024   // h*w = 32*32
#define DIM   512
#define SCALE 0.044194173824159216f   // 512^-0.5

// ---------------------------------------------------------------------------
// Kernel 0: convert the 4 weight matrices to f16, zero g accumulator.
// Re-run every launch (graph-safe, deterministic).
// ---------------------------------------------------------------------------
__global__ __launch_bounds__(256) void convert_kernel(
    const float* __restrict__ Wq, const float* __restrict__ Wk,
    const float* __restrict__ Wp, const float* __restrict__ Wf,
    _Float16* __restrict__ q16, _Float16* __restrict__ k16,
    _Float16* __restrict__ p16, _Float16* __restrict__ f16o,
    float* __restrict__ g)
{
    int i = blockIdx.x * 256 + threadIdx.x;
    if (i < DIM * DIM) {
        q16[i] = (_Float16)Wq[i];
        k16[i] = (_Float16)Wk[i];
        p16[i] = (_Float16)Wp[i];
        f16o[i] = (_Float16)Wf[i];
    }
    if (i < BATCH * DIM) g[i] = 0.0f;
}

// ---------------------------------------------------------------------------
// A-fragment loader: 16x32 f16 A tile, contiguous k order (k = 16*hid + s), same as B.
// lane<16  holds row M=lane,    halves = K[0..7]  then K[8..15]
// lane>=16 holds row M=lane-16, halves = K[16..23] then K[24..31]
// ---------------------------------------------------------------------------
__device__ inline v16h load_a_frag(const _Float16* row, int k0, int hid)
{
    v8h lo = *(const v8h*)(row + k0 + hid * 16);
    v8h hi = *(const v8h*)(row + k0 + hid * 16 + 8);
    return __builtin_shufflevector(lo, hi, 0,1,2,3,4,5,6,7,8,9,10,11,12,13,14,15);
}

// ---------------------------------------------------------------------------
// Kernel 1: Q = l2norm(X Wq^T + bq), K = l2norm(X Wk^T + bk), qw = Q . w_g
// grid = (NTOK/16, BATCH), 256 threads (8 waves).
// Waves 0..3 -> Q (128 d-cols each), waves 4..7 -> K.
// ---------------------------------------------------------------------------
__global__ __launch_bounds__(256) void qk_kernel(
    const float* __restrict__ x,          // [B][512][1024]
    const _Float16* __restrict__ Wq16,
    const float* __restrict__ bq,
    const _Float16* __restrict__ Wk16,
    const float* __restrict__ bk,
    const float* __restrict__ wg,         // [512]
    float* __restrict__ Qn,               // [B][N][D] normalized, f32
    _Float16* __restrict__ Kn,            // [B][N][D] normalized, f16
    float* __restrict__ qw)               // [B][N]
{
    __shared__ _Float16 Xs[16][520];      // A tile, f16, padded pitch
    __shared__ float    Qs[16][516];
    __shared__ float    Ks[16][516];
    __shared__ float    invn[2][16];
    __shared__ float    qwsh[16];

    const int ntile = blockIdx.x;
    const int b     = blockIdx.y;
    const int tid   = threadIdx.x;

    // Stage X^T tile: X[n][c] lives at x[b][c][n] -> transpose into Xs[n-local][c]
    const float* xb = x + (size_t)b * DIM * NTOK + (size_t)ntile * 16;
    for (int idx = tid; idx < 16 * DIM; idx += 256) {
        int p = idx & 15;         // token within tile (coalesced over p)
        int c = idx >> 4;
        Xs[p][c] = (_Float16)xb[(size_t)c * NTOK + p];
    }
    __syncthreads();

    const int wave = tid >> 5;
    const int lane = tid & 31;
    const int m    = lane & 15;
    const int hid  = lane >> 4;
    const bool isK = wave >= 4;
    const _Float16* Wm = isK ? Wk16 : Wq16;
    const int dbase = (wave & 3) * 128;   // 8 tiles of 16 columns each

    v8f acc[8] = {};
    const _Float16* arow = &Xs[m][0];
    for (int k0 = 0; k0 < DIM; k0 += 32) {
        v16h a = load_a_frag(arow, k0, hid);
        #pragma unroll
        for (int i = 0; i < 8; ++i) {
            const int d0 = dbase + i * 16;
            // B frag: lane = column n=m, K range hid*16..+15 -> contiguous weight row
            v16h bfrag = *(const v16h*)(Wm + (size_t)(d0 + m) * DIM + k0 + hid * 16);
            acc[i] = __builtin_amdgcn_wmma_f32_16x16x32_f16(
                false, a, false, bfrag, (short)0, acc[i], false, false);
        }
    }

    // Epilogue: +bias into LDS (D-frag: row = r + 8*hid, col = d0 + m)
    const float* bias = isK ? bk : bq;
    float (*Os)[516] = isK ? Ks : Qs;
    #pragma unroll
    for (int i = 0; i < 8; ++i) {
        const int col = dbase + i * 16 + m;
        const float bv = bias[col];
        #pragma unroll
        for (int r = 0; r < 8; ++r)
            Os[r + 8 * hid][col] = acc[i][r] + bv;
    }
    __syncthreads();

    // Row l2 norms (+ qw = dot(normalized q, w_g)); one thread per (mat,row)
    if (tid < 32) {
        const int row = tid & 15;
        const bool kk = tid >= 16;
        const float (*S)[516] = kk ? Ks : Qs;
        float ss = 0.0f, qd = 0.0f;
        for (int i = 0; i < DIM; ++i) {
            float v = S[row][i];
            ss += v * v;
            if (!kk) qd += v * wg[i];
        }
        float inv = 1.0f / fmaxf(sqrtf(ss), 1e-12f);
        invn[kk ? 1 : 0][row] = inv;
        if (!kk) qwsh[row] = qd * inv;
    }
    __syncthreads();

    // Store normalized Q (f32) and K (f16)
    const size_t base = ((size_t)b * NTOK + (size_t)ntile * 16) * DIM;
    for (int idx = tid; idx < 16 * DIM; idx += 256) {
        int row = idx >> 9, d = idx & 511;
        Qn[base + (size_t)row * DIM + d] = Qs[row][d] * invn[0][row];
        Kn[base + (size_t)row * DIM + d] = (_Float16)(Ks[row][d] * invn[1][row]);
    }
    if (tid < 16) qw[(size_t)b * NTOK + ntile * 16 + tid] = qwsh[tid];
}

// ---------------------------------------------------------------------------
// Kernel 2: a = l2norm(qw*scale over n); g[d] = sum_n a_n Q[n,d]
// grid = (BATCH, 4): each block reduces a 256-token chunk, atomicAdd into g.
// ---------------------------------------------------------------------------
__global__ __launch_bounds__(256) void ctx_kernel(
    const float* __restrict__ Qn, const float* __restrict__ qw,
    float* __restrict__ g)
{
    __shared__ float red[256];
    __shared__ float qws[256];
    const int b = blockIdx.x, chunk = blockIdx.y, tid = threadIdx.x;

    const float* qwb = qw + (size_t)b * NTOK;
    float ss = 0.0f;
    for (int i = tid; i < NTOK; i += 256) { float v = qwb[i]; ss += v * v; }
    red[tid] = ss;
    qws[tid] = qwb[chunk * 256 + tid];
    __syncthreads();
    for (int s = 128; s > 0; s >>= 1) {
        if (tid < s) red[tid] += red[tid + s];
        __syncthreads();
    }
    // a = qw*scale / max(||qw*scale||, eps)  ->  coef applied to the qw-weighted sum
    const float coef = SCALE / fmaxf(sqrtf(red[0]) * SCALE, 1e-12f);

    const float* Qb = Qn + ((size_t)b * NTOK + (size_t)chunk * 256) * DIM;
    float a0 = 0.0f, a1 = 0.0f;
    const int d0 = tid, d1 = tid + 256;
    for (int n = 0; n < 256; ++n) {
        float w = qws[n];
        a0 += w * Qb[(size_t)n * DIM + d0];
        a1 += w * Qb[(size_t)n * DIM + d1];
    }
    atomicAdd(&g[(size_t)b * DIM + d0], a0 * coef);
    atomicAdd(&g[(size_t)b * DIM + d1], a1 * coef);
}

// ---------------------------------------------------------------------------
// Kernel 3: out1 = (g (*) K) Wp^T + bp + Q ; out = out1 Wf^T + bf ; transpose.
// grid = (NTOK/16, BATCH), 256 threads (8 waves x 4 tiles per GEMM).
// ---------------------------------------------------------------------------
__global__ __launch_bounds__(256) void out_kernel(
    const _Float16* __restrict__ Kn, const float* __restrict__ Qn,
    const float* __restrict__ g,
    const _Float16* __restrict__ Wp16, const float* __restrict__ bp,
    const _Float16* __restrict__ Wf16, const float* __restrict__ bf,
    float* __restrict__ out)                 // [B][512][1024]
{
    __shared__ _Float16 Ys[16][520];
    __shared__ _Float16 O1s[16][520];
    __shared__ float    O2s[16][516];

    const int ntile = blockIdx.x;
    const int b     = blockIdx.y;
    const int tid   = threadIdx.x;
    const size_t tb = ((size_t)b * NTOK + (size_t)ntile * 16) * DIM;
    const float* gb = g + (size_t)b * DIM;

    // Stage A1 = g (*) K as f16
    for (int idx = tid; idx < 16 * DIM; idx += 256) {
        int row = idx >> 9, d = idx & 511;
        Ys[row][d] = (_Float16)((float)Kn[tb + (size_t)row * DIM + d] * gb[d]);
    }
    __syncthreads();

    const int wave = tid >> 5;
    const int lane = tid & 31;
    const int m    = lane & 15;
    const int hid  = lane >> 4;

    // GEMM 1: vs Wp
    v8f acc1[4] = {};
    for (int k0 = 0; k0 < DIM; k0 += 32) {
        v16h a = load_a_frag(&Ys[m][0], k0, hid);
        #pragma unroll
        for (int i = 0; i < 4; ++i) {
            const int e0 = wave * 64 + i * 16;
            v16h bfrag = *(const v16h*)(Wp16 + (size_t)(e0 + m) * DIM + k0 + hid * 16);
            acc1[i] = __builtin_amdgcn_wmma_f32_16x16x32_f16(
                false, a, false, bfrag, (short)0, acc1[i], false, false);
        }
    }
    // epilogue: + bp + Q residual -> f16 staging for GEMM 2
    #pragma unroll
    for (int i = 0; i < 4; ++i) {
        const int col = wave * 64 + i * 16 + m;
        const float bv = bp[col];
        #pragma unroll
        for (int r = 0; r < 8; ++r) {
            const int row = r + 8 * hid;
            O1s[row][col] = (_Float16)(acc1[i][r] + bv + Qn[tb + (size_t)row * DIM + col]);
        }
    }
    __syncthreads();

    // GEMM 2: vs Wf
    v8f acc2[4] = {};
    for (int k0 = 0; k0 < DIM; k0 += 32) {
        v16h a = load_a_frag(&O1s[m][0], k0, hid);
        #pragma unroll
        for (int i = 0; i < 4; ++i) {
            const int e0 = wave * 64 + i * 16;
            v16h bfrag = *(const v16h*)(Wf16 + (size_t)(e0 + m) * DIM + k0 + hid * 16);
            acc2[i] = __builtin_amdgcn_wmma_f32_16x16x32_f16(
                false, a, false, bfrag, (short)0, acc2[i], false, false);
        }
    }
    #pragma unroll
    for (int i = 0; i < 4; ++i) {
        const int col = wave * 64 + i * 16 + m;
        const float bv = bf[col];
        #pragma unroll
        for (int r = 0; r < 8; ++r)
            O2s[r + 8 * hid][col] = acc2[i][r] + bv;
    }
    __syncthreads();

    // Transposed, coalesced store: out[b][c][p] ; 4 threads x float4 per c-row
    float* ob = out + (size_t)b * DIM * NTOK + (size_t)ntile * 16;
    const int pofs = (tid & 3) * 4;
    for (int c = tid >> 2; c < DIM; c += 64) {
        float4 v;
        v.x = O2s[pofs + 0][c];
        v.y = O2s[pofs + 1][c];
        v.z = O2s[pofs + 2][c];
        v.w = O2s[pofs + 3][c];
        *(float4*)(ob + (size_t)c * NTOK + pofs) = v;
    }
}

// ---------------------------------------------------------------------------
extern "C" void kernel_launch(void* const* d_in, const int* in_sizes, int n_in,
                              void* d_out, int out_size, void* d_ws, size_t ws_size,
                              hipStream_t stream) {
    const float* x_4d = (const float*)d_in[0];
    const float* Wq   = (const float*)d_in[1];
    const float* bq   = (const float*)d_in[2];
    const float* Wk   = (const float*)d_in[3];
    const float* bk   = (const float*)d_in[4];
    const float* w_g  = (const float*)d_in[5];
    const float* Wp   = (const float*)d_in[6];
    const float* bp   = (const float*)d_in[7];
    const float* Wf   = (const float*)d_in[8];
    const float* bf   = (const float*)d_in[9];
    float* out = (float*)d_out;

    // Workspace layout
    char* ws = (char*)d_ws;
    const size_t WBYTES = (size_t)DIM * DIM * sizeof(_Float16);   // 512 KB each
    _Float16* Wq16 = (_Float16*)(ws);
    _Float16* Wk16 = (_Float16*)(ws + 1 * WBYTES);
    _Float16* Wp16 = (_Float16*)(ws + 2 * WBYTES);
    _Float16* Wf16 = (_Float16*)(ws + 3 * WBYTES);
    char* p = ws + 4 * WBYTES;
    float*    Qn = (float*)p;    p += (size_t)BATCH * NTOK * DIM * sizeof(float);    // 64 MB
    _Float16* Kn = (_Float16*)p; p += (size_t)BATCH * NTOK * DIM * sizeof(_Float16); // 32 MB
    float*    qw = (float*)p;    p += (size_t)BATCH * NTOK * sizeof(float);
    float*    g  = (float*)p;

    convert_kernel<<<dim3((DIM * DIM + 255) / 256), dim3(256), 0, stream>>>(
        Wq, Wk, Wp, Wf, Wq16, Wk16, Wp16, Wf16, g);

    qk_kernel<<<dim3(NTOK / 16, BATCH), dim3(256), 0, stream>>>(
        x_4d, Wq16, bq, Wk16, bk, w_g, Qn, Kn, qw);

    ctx_kernel<<<dim3(BATCH, 4), dim3(256), 0, stream>>>(Qn, qw, g);

    out_kernel<<<dim3(NTOK / 16, BATCH), dim3(256), 0, stream>>>(
        Kn, Qn, g, Wp16, bp, Wf16, bf, out);
}
